// attention_module_multi_head_RN_cls_59236188947001
// MI455X (gfx1250) — compile-verified, matrix operand order checked by reading, not run on hardware
//
#include <hip/hip_runtime.h>
#include <hip/hip_bf16.h>

// ---------------------------------------------------------------------------
// MI455X (gfx1250) relation-network attention head.
//
//  * Algebraic fusion:  einsum('igj,jd->igd') + einsum('igd,gkd->igk')
//    == sm[i,g,:] @ P[g] with P[g] = roi @ W_conv[g]^T  (~45G -> ~13G FLOPs).
//  * All GEMMs on v_wmma_f32_16x16x32_bf16 (bf16 in, f32 accumulate).
//  * Ping-pong (2-stage) software pipelining: two named fragment buffer sets
//    alternate between "being loaded" and "feeding WMMA", so there are no
//    buffer-rotation v_mov copies and waits are partial (s_wait_loadcnt N).
//  * Attention fused per (16-row tile, group): the 64KB positional log-prior
//    strip is bulk-copied global->LDS with GLOBAL_LOAD_ASYNC_TO_LDS_B128
//    (ASYNCcnt, no VGPR round-trip), overlapped with the QK^T WMMAs and fenced
//    with s_wait_asynccnt; exact two-pass softmax via wave32 shuffles with
//    exp() values written back to the LDS strip for reuse by the PV pass.
//  * Only column sums of the residual feed the final 80-wide sigmoid head.
// ---------------------------------------------------------------------------

typedef __attribute__((ext_vector_type(8)))  __bf16 v8bf;
typedef __attribute__((ext_vector_type(16))) __bf16 v16bf;
typedef __attribute__((ext_vector_type(8)))  float  v8f;

#define N_TOK 1024
#define FEAT  1024
#define G     16
#define DG    64
#define CLS   80

// ---- WMMA fragment helpers (CDNA5 16-bit A/B layout: lane=row(col),
// half-wave selects K-octets; two 16B loads per lane) -------------------------
__device__ __forceinline__ v16bf load_tile16x32(const __bf16* base, int ld) {
  const int lane = threadIdx.x & 31;
  const int r    = lane & 15;   // row (A) / column (B)
  const int h    = lane >> 4;   // half-wave selects K-octet
  const __bf16* p = base + (size_t)r * ld + 8 * h;
  v8bf lo = *(const v8bf*)(p);        // K = 8h .. 8h+7
  v8bf hi = *(const v8bf*)(p + 16);   // K = 16+8h .. 16+8h+7
  v16bf a;
#pragma unroll
  for (int e = 0; e < 8; ++e) { a[e] = lo[e]; a[8 + e] = hi[e]; }
  return a;
}

__device__ __forceinline__ v8f wmma_bf16(v16bf a, v16bf b, v8f c) {
  // 8 args: (neg_a, A, neg_b, B, c_mod, C, reuse_a, reuse_b)
  return __builtin_amdgcn_wmma_f32_16x16x32_bf16(false, a, false, b,
                                                 (short)0, c, false, false);
}

// ---- precision-conversion / layout kernels ---------------------------------
__global__ void k_f32_to_bf16(const float* __restrict__ in, __bf16* __restrict__ out, int n) {
  int i = blockIdx.x * blockDim.x + threadIdx.x;
  if (i < n) out[i] = (__bf16)in[i];
}

// in: rows x cols (f32, row-major)  ->  out: cols x rows (bf16)
__global__ void k_transpose_to_bf16(const float* __restrict__ in, __bf16* __restrict__ out,
                                    int rows, int cols) {
  int i = blockIdx.x * blockDim.x + threadIdx.x;
  if (i < rows * cols) {
    int r = i / cols, c = i - r * cols;
    out[(size_t)c * rows + r] = (__bf16)in[i];
  }
}

// ---- generic bf16 WMMA GEMM: C[MxN] = A[MxK] @ Bc^T (+bias), bf16 out ------
// A row-major; Bc is N x K row-major (B's columns contiguous in K).
// Block = 8 waves; each wave owns a 16x64 C tile. K is assumed a multiple of
// 64; the K-loop is a 2-stage ping-pong (no buffer-rotation moves).
__global__ void __launch_bounds__(256)
k_gemm_bf16(const __bf16* __restrict__ A, const __bf16* __restrict__ Bc,
            const float* __restrict__ bias, __bf16* __restrict__ C,
            int M, int Nn, int K) {
  const int wave = threadIdx.x >> 5;
  const int lane = threadIdx.x & 31;
  const int m0 = blockIdx.x * 16;
  const int n0 = (blockIdx.y * 8 + wave) * 64;
  const __bf16* Ap = A + (size_t)m0 * K;

  v8f acc[4] = {};
  v16bf aA = load_tile16x32(Ap, K);
  v16bf bA[4];
#pragma unroll
  for (int t = 0; t < 4; ++t)
    bA[t] = load_tile16x32(Bc + (size_t)(n0 + t * 16) * K, K);

#pragma unroll 1
  for (int k0 = 0; k0 < K; k0 += 64) {
    // stage B loads (k0+32) while stage A computes
    v16bf aB = load_tile16x32(Ap + k0 + 32, K);
    v16bf bB[4];
#pragma unroll
    for (int t = 0; t < 4; ++t)
      bB[t] = load_tile16x32(Bc + (size_t)(n0 + t * 16) * K + k0 + 32, K);
#pragma unroll
    for (int t = 0; t < 4; ++t) acc[t] = wmma_bf16(aA, bA[t], acc[t]);

    // stage A loads (k0+64) while stage B computes
    if (k0 + 64 < K) {
      aA = load_tile16x32(Ap + k0 + 64, K);
#pragma unroll
      for (int t = 0; t < 4; ++t)
        bA[t] = load_tile16x32(Bc + (size_t)(n0 + t * 16) * K + k0 + 64, K);
    }
#pragma unroll
    for (int t = 0; t < 4; ++t) acc[t] = wmma_bf16(aB, bB[t], acc[t]);
  }

  const int h = lane >> 4, n = lane & 15;
#pragma unroll
  for (int t = 0; t < 4; ++t)
#pragma unroll
    for (int v = 0; v < 8; ++v) {
      int m  = v + 8 * h;                 // C layout: VGPR v -> row v / v+8
      int nc = n0 + t * 16 + n;
      float val = acc[t][v] + (bias ? bias[nc] : 0.0f);
      C[(size_t)(m0 + m) * Nn + nc] = (__bf16)val;
    }
}

// ---- positional log-prior:  logw[g][i][j] = log(max(relu(emb.Wpos+b), 1e-6))
__device__ __forceinline__ float logclamp(float d) {
  return __logf(fmaxf(fabsf(d), 1e-3f));
}

__global__ void __launch_bounds__(256)
k_logw(const float* __restrict__ bbox, const float* __restrict__ W_pos,
       const float* __restrict__ b_pos, float* __restrict__ logw) {
  __shared__ float sW[64 * 16];
  __shared__ float sB[16];
  for (int t = threadIdx.x; t < 64 * 16; t += blockDim.x) sW[t] = W_pos[t];
  if (threadIdx.x < 16) sB[threadIdx.x] = b_pos[threadIdx.x];
  __syncthreads();

  const int idx = blockIdx.x * blockDim.x + threadIdx.x;   // i*N + j
  const int i = idx >> 10, j = idx & (N_TOK - 1);
  const float xi0 = bbox[i * 4 + 0], yi0 = bbox[i * 4 + 1];
  const float xi1 = bbox[i * 4 + 2], yi1 = bbox[i * 4 + 3];
  const float xj0 = bbox[j * 4 + 0], yj0 = bbox[j * 4 + 1];
  const float xj1 = bbox[j * 4 + 2], yj1 = bbox[j * 4 + 3];
  const float wi = xi1 - xi0 + 1.0f, hi = yi1 - yi0 + 1.0f;
  const float wj = xj1 - xj0 + 1.0f, hj = yj1 - yj0 + 1.0f;
  const float cxi = 0.5f * (xi0 + xi1), cyi = 0.5f * (yi0 + yi1);
  const float cxj = 0.5f * (xj0 + xj1), cyj = 0.5f * (yj0 + yj1);

  float pos[4];
  pos[0] = logclamp((cxi - cxj) / wi);
  pos[1] = logclamp((cyi - cyj) / hi);
  pos[2] = logclamp(wi / wj);
  pos[3] = logclamp(hi / hj);

  float emb[64];
#pragma unroll
  for (int c = 0; c < 4; ++c)
#pragma unroll
    for (int t = 0; t < 8; ++t) {
      // 1000^(t/8) frequencies; invdim = exp(-t*ln(1000)/8)
      float a = pos[c] * __expf(-(float)t * 0.8634736f);
      float s, co;
      __sincosf(a, &s, &co);
      emb[c * 16 + t]     = s;
      emb[c * 16 + 8 + t] = co;
    }
#pragma unroll 1
  for (int g = 0; g < G; ++g) {
    float acc = sB[g];
#pragma unroll
    for (int e = 0; e < 64; ++e) acc += emb[e] * sW[e * 16 + g];
    acc = fmaxf(acc, 0.0f);        // relu
    acc = fmaxf(acc, 1e-6f);       // EPS_AFF clamp
    logw[((size_t)g << 20) + idx] = __logf(acc);
  }
}

// ---- fused attention: async logw->LDS + QK^T WMMA -> softmax -> PV WMMA ----
// One wave per (16-row i-tile, group g).  LDS: 16x1024 f32 strip = 64KB.
__global__ void __launch_bounds__(32)
k_attn(const __bf16* __restrict__ qB, const __bf16* __restrict__ kB,
       const __bf16* __restrict__ PT, const float* __restrict__ logw,
       const float* __restrict__ b_conv, float* __restrict__ lin) {
  __shared__ float s_sc[16 * N_TOK];
  const int i0   = blockIdx.x * 16;
  const int g    = blockIdx.y;
  const int lane = threadIdx.x & 31;
  const int h = lane >> 4, n = lane & 15;
  const int gofs = g * DG;
  const float* lw = logw + ((size_t)g << 20);

  // Prologue: bulk-copy the contiguous 64KB logw strip logw[g][i0..i0+15][*]
  // straight into LDS with the CDNA5 async path (no VGPR round-trip).
  // 32 lanes x 16B = 512B per instruction, 128 instructions total.
  {
    uint32_t lbase = (uint32_t)(size_t)(&s_sc[0]);   // low 32 bits = LDS offset
    const char* gsrc = (const char*)(lw + (size_t)i0 * N_TOK);
#pragma unroll 1
    for (int t = 0; t < 128; ++t) {
      uint32_t off = (uint32_t)(t * 512 + lane * 16);
      asm volatile("global_load_async_to_lds_b128 %0, %1, off"
                   :: "v"(lbase + off), "v"(gsrc + off)
                   : "memory");
    }
  }

  // Pass 1: S = (q k^T) * rsqrt(DG) accumulated onto the LDS prior strip.
  // q fragments are loop-invariant (hoisted); k fragments ping-pong in a
  // 2-stage pipeline (j-step 32 = two 16-wide score tiles per iteration).
  const __bf16* qp = qB + (size_t)i0 * FEAT + gofs;
  v16bf aq0 = load_tile16x32(qp, FEAT);
  v16bf aq1 = load_tile16x32(qp + 32, FEAT);
  v16bf kfA0 = load_tile16x32(kB + gofs, FEAT);
  v16bf kfA1 = load_tile16x32(kB + gofs + 32, FEAT);
#pragma unroll 1
  for (int j0 = 0; j0 < N_TOK; j0 += 32) {
    // stage B loads (columns j0+16..j0+31) while stage A computes
    const __bf16* kpB = kB + (size_t)(j0 + 16) * FEAT + gofs;
    v16bf kfB0 = load_tile16x32(kpB, FEAT);
    v16bf kfB1 = load_tile16x32(kpB + 32, FEAT);
    v8f c = {};
    c = wmma_bf16(aq0, kfA0, c);
    c = wmma_bf16(aq1, kfA1, c);
    if (j0 == 0)  // fence the async prior copy before first LDS accumulate
      asm volatile("s_wait_asynccnt 0x0" ::: "memory");
#pragma unroll
    for (int v = 0; v < 8; ++v) {
      int m = v + 8 * h;
      s_sc[m * N_TOK + j0 + n] += 0.125f * c[v];
    }
    // stage A loads (columns j0+32..) while stage B computes
    if (j0 + 32 < N_TOK) {
      const __bf16* kpA = kB + (size_t)(j0 + 32) * FEAT + gofs;
      kfA0 = load_tile16x32(kpA, FEAT);
      kfA1 = load_tile16x32(kpA + 32, FEAT);
    }
    v8f c2 = {};
    c2 = wmma_bf16(aq0, kfB0, c2);
    c2 = wmma_bf16(aq1, kfB1, c2);
#pragma unroll
    for (int v = 0; v < 8; ++v) {
      int m = v + 8 * h;
      s_sc[m * N_TOK + j0 + 16 + n] += 0.125f * c2[v];
    }
  }
  __syncthreads();

  // Pass 2: exact row max / sum (lane pair per row, wave32 shuffle combine);
  // exp(s - max) is written back into the strip for reuse by pass 3.
  const int row = n;
  const int jbeg = h * 512, jend = h * 512 + 512;
  float mx = -3.0e38f;
#pragma unroll 1
  for (int j = jbeg; j < jend; ++j) mx = fmaxf(mx, s_sc[row * N_TOK + j]);
  mx = fmaxf(mx, __shfl_xor(mx, 16, 32));
  float sum = 0.0f;
#pragma unroll 1
  for (int j = jbeg; j < jend; ++j) {
    float e = __expf(s_sc[row * N_TOK + j] - mx);
    s_sc[row * N_TOK + j] = e;
    sum += e;
  }
  sum += __shfl_xor(sum, 16, 32);
  const float rinv = 1.0f / sum;
  __syncthreads();

  // Pass 3: lin[i,g,:] = softmax_row @ P[g]  (16 x 64 x 1024 WMMA),
  // P fragments ping-pong (j-step 64 = two K32 stages per iteration).
  const __bf16* Pg = PT + (size_t)gofs * N_TOK;
  v8f acc[4] = {};
  v16bf pA[4];
#pragma unroll
  for (int t = 0; t < 4; ++t)
    pA[t] = load_tile16x32(Pg + (size_t)(t * 16) * N_TOK, N_TOK);
#pragma unroll 1
  for (int jt = 0; jt < N_TOK; jt += 64) {
    // stage B loads (jt+32) while stage A computes
    v16bf pB[4];
#pragma unroll
    for (int t = 0; t < 4; ++t)
      pB[t] = load_tile16x32(Pg + (size_t)(t * 16) * N_TOK + jt + 32, N_TOK);
    v16bf a;
#pragma unroll
    for (int e = 0; e < 8; ++e) {
      a[e]     = (__bf16)(s_sc[row * N_TOK + jt + 8 * h + e] * rinv);
      a[8 + e] = (__bf16)(s_sc[row * N_TOK + jt + 16 + 8 * h + e] * rinv);
    }
#pragma unroll
    for (int t = 0; t < 4; ++t) acc[t] = wmma_bf16(a, pA[t], acc[t]);

    // stage A loads (jt+64) while stage B computes
    if (jt + 64 < N_TOK) {
#pragma unroll
      for (int t = 0; t < 4; ++t)
        pA[t] = load_tile16x32(Pg + (size_t)(t * 16) * N_TOK + jt + 64, N_TOK);
    }
    v16bf a2;
#pragma unroll
    for (int e = 0; e < 8; ++e) {
      a2[e]     = (__bf16)(s_sc[row * N_TOK + jt + 32 + 8 * h + e] * rinv);
      a2[8 + e] = (__bf16)(s_sc[row * N_TOK + jt + 48 + 8 * h + e] * rinv);
    }
#pragma unroll
    for (int t = 0; t < 4; ++t) acc[t] = wmma_bf16(a2, pB[t], acc[t]);
  }
#pragma unroll
  for (int t = 0; t < 4; ++t)
#pragma unroll
    for (int v = 0; v < 8; ++v) {
      int m = v + 8 * h;
      int f = gofs + t * 16 + n;
      lin[(size_t)(i0 + m) * FEAT + f] = acc[t][v] + b_conv[f];
    }
}

// ---- final reductions: colsum of (roi + lin), then 1024x80 matvec+sigmoid --
__global__ void k_colsum(const float* __restrict__ roi, const float* __restrict__ lin,
                         float* __restrict__ colsum) {
  int f = blockIdx.x * blockDim.x + threadIdx.x;
  if (f >= FEAT) return;
  float s = 0.0f;
  for (int i = 0; i < N_TOK; ++i)
    s += roi[(size_t)i * FEAT + f] + lin[(size_t)i * FEAT + f];
  colsum[f] = s;
}

__global__ void k_logits(const float* __restrict__ colsum, const float* __restrict__ W_ro,
                         const float* __restrict__ b_ro, float* __restrict__ out) {
  int c = blockIdx.x * blockDim.x + threadIdx.x;
  if (c >= CLS) return;
  float acc = b_ro[c];
  for (int k = 0; k < FEAT; ++k) acc += colsum[k] * W_ro[k * CLS + c];
  out[c] = 1.0f / (1.0f + __expf(-acc));
}

// ---------------------------------------------------------------------------
extern "C" void kernel_launch(void* const* d_in, const int* in_sizes, int n_in,
                              void* d_out, int out_size, void* d_ws, size_t ws_size,
                              hipStream_t stream) {
  const float* bbox   = (const float*)d_in[0];
  const float* roi    = (const float*)d_in[1];
  const float* W_pos  = (const float*)d_in[2];
  const float* b_pos  = (const float*)d_in[3];
  const float* Wq     = (const float*)d_in[4];
  const float* bq     = (const float*)d_in[5];
  const float* Wk     = (const float*)d_in[6];
  const float* bk     = (const float*)d_in[7];
  const float* W_conv = (const float*)d_in[8];
  const float* b_conv = (const float*)d_in[9];
  const float* W_ro   = (const float*)d_in[10];
  const float* b_ro   = (const float*)d_in[11];

  char* ws = (char*)d_ws;
  const size_t MB = 1024ull * 1024ull;
  __bf16* roiB   = (__bf16*)(ws + 0 * MB);    // 2MB  roi_feat bf16 (N,FEAT)
  __bf16* WqT    = (__bf16*)(ws + 2 * MB);    // 2MB  Wq^T bf16
  __bf16* WkT    = (__bf16*)(ws + 4 * MB);    // 2MB  Wk^T bf16
  __bf16* WcB    = (__bf16*)(ws + 6 * MB);    // 2MB  W_conv flat bf16 (G*DG, FEAT)
  __bf16* qB     = (__bf16*)(ws + 8 * MB);    // 2MB  q bf16 (N, G*DG)
  __bf16* kB     = (__bf16*)(ws + 10 * MB);   // 2MB  k bf16 (N, G*DG)
  __bf16* PT     = (__bf16*)(ws + 12 * MB);   // 2MB  P^T bf16 (G*DG, N)
  float*  lin    = (float*) (ws + 14 * MB);   // 4MB  lin f32 (N, FEAT)
  float*  colsum = (float*) (ws + 18 * MB);   // 4KB
  float*  logw   = (float*) (ws + 19 * MB);   // 64MB (G, N, N) f32

  const int MELT = N_TOK * FEAT;  // 1M elements

  // Precision conversion / layout staging
  k_f32_to_bf16<<<MELT / 256, 256, 0, stream>>>(roi, roiB, MELT);
  k_f32_to_bf16<<<MELT / 256, 256, 0, stream>>>(W_conv, WcB, MELT);
  k_transpose_to_bf16<<<MELT / 256, 256, 0, stream>>>(Wq, WqT, FEAT, FEAT);
  k_transpose_to_bf16<<<MELT / 256, 256, 0, stream>>>(Wk, WkT, FEAT, FEAT);

  // Projections + fused-conv precompute (three 1024^3 bf16 WMMA GEMMs)
  k_gemm_bf16<<<dim3(64, 2), 256, 0, stream>>>(roiB, WqT, bq, qB, N_TOK, FEAT, FEAT);
  k_gemm_bf16<<<dim3(64, 2), 256, 0, stream>>>(roiB, WkT, bk, kB, N_TOK, FEAT, FEAT);
  k_gemm_bf16<<<dim3(64, 2), 256, 0, stream>>>(WcB, roiB, nullptr, PT, G * DG, N_TOK, FEAT);

  // Positional log-prior table (G,N,N)
  k_logw<<<(N_TOK * N_TOK) / 256, 256, 0, stream>>>(bbox, W_pos, b_pos, logw);

  // Fused attention (async prior copy + scores -> softmax -> PV)
  k_attn<<<dim3(N_TOK / 16, G), 32, 0, stream>>>(qB, kB, PT, logw, b_conv, lin);

  // Final reduction to 80 sigmoid logits
  k_colsum<<<FEAT / 256, 256, 0, stream>>>(roi, lin, colsum);
  k_logits<<<1, 128, 0, stream>>>(colsum, W_ro, b_ro, (float*)d_out);
}
